// AlignMem_8546984919617
// MI455X (gfx1250) — matrix operand order repaired and checked by reading, not compiled, not run on hardware
//
#include <hip/hip_runtime.h>
#include <math.h>

#define NCLS 201
#define DIMC 2048
#define SSEL 32
#define BSZ  128
#define HWP  196   // 14*14

typedef __attribute__((ext_vector_type(16))) __bf16 v16bf;
typedef __attribute__((ext_vector_type(8)))  float  v8f;

// ---------- bf16 helpers (RNE) for bf16x3 split-precision WMMA ----------
__device__ __forceinline__ __bf16 f2bf(float x) {
    unsigned u = __builtin_bit_cast(unsigned, x);
    unsigned r = u + 0x7FFFu + ((u >> 16) & 1u);
    return __builtin_bit_cast(__bf16, (unsigned short)(r >> 16));
}
__device__ __forceinline__ float bf2f(__bf16 h) {
    unsigned u = ((unsigned)__builtin_bit_cast(unsigned short, h)) << 16;
    return __builtin_bit_cast(float, u);
}

// ---------- 1) softmax stats + masks (one block per batch row) ----------
__global__ void k_scores(const float* __restrict__ scores, const float* __restrict__ bconf,
                         const float* __restrict__ ctx, const int* __restrict__ labels,
                         int* pred_pos, float* pred_val, int* updm, int* fwdm, int* errm) {
    const int b = blockIdx.x, t = threadIdx.x;
    __shared__ float sv[256];
    __shared__ int   si[256];
    const int l = labels[b];
    float s = (t < NCLS) ? scores[b * NCLS + t] : -INFINITY;
    sv[t] = s; si[t] = t; __syncthreads();
    for (int off = 128; off > 0; off >>= 1) {
        if (t < off) {
            float v2 = sv[t + off]; int i2 = si[t + off];
            if (v2 > sv[t] || (v2 == sv[t] && i2 < si[t])) { sv[t] = v2; si[t] = i2; }
        }
        __syncthreads();
    }
    const float smax = sv[0]; const int pp = si[0];
    __syncthreads();
    sv[t] = (t < NCLS) ? expf(s - smax) : 0.f; __syncthreads();
    for (int off = 128; off > 0; off >>= 1) { if (t < off) sv[t] += sv[t + off]; __syncthreads(); }
    const float pv = 1.f / sv[0];             // max softmax prob
    __syncthreads();
    float acc = 0.f;
    for (int i = t; i < DIMC; i += 256) acc += ctx[(size_t)l * DIMC + i];
    sv[t] = acc; __syncthreads();
    for (int off = 128; off > 0; off >>= 1) { if (t < off) sv[t] += sv[t + off]; __syncthreads(); }
    if (t == 0) {
        pred_pos[b] = pp; pred_val[b] = pv;
        const float lc = bconf[l];
        const bool correct = (pp == l);
        const bool bg = (l != NCLS) || (pp != NCLS);
        updm[b] = (correct && (pv - lc > 0.1f) && bg) ? 1 : 0;
        fwdm[b] = (correct && (lc - pv > 0.1f) && bg && (lc != 0.f)) ? 1 : 0;
        errm[b] = (!correct && (sv[0] != 0.f)) ? 1 : 0;
    }
}

// ---------- 2) hm partials: grid (chunk=8, b=128), deterministic ----------
__global__ void k_hm_part(const float* __restrict__ feat, float* __restrict__ hmp) {
    const int chunk = blockIdx.x, b = blockIdx.y, t = threadIdx.x;
    if (t >= HWP) return;
    const float* f = feat + ((size_t)b * DIMC + chunk * 256) * HWP + t;
    float acc = 0.f;
    for (int d = 0; d < 256; ++d) acc += fmaxf(f[d * HWP], 0.f);
    hmp[(b * 8 + chunk) * HWP + t] = acc;
}

// ---------- 3) reduce + column-normalize + exact top-32 (bitonic) ----------
__global__ void k_topk(const float* __restrict__ hmp, float* __restrict__ pick_val,
                       int* __restrict__ pick_pos) {
    const int b = blockIdx.x, t = threadIdx.x;  // 256 threads
    __shared__ float sv[256];
    __shared__ int   si[256];
    __shared__ float nrm[14];
    float v = 0.f;
    if (t < HWP) {
#pragma unroll
        for (int c = 0; c < 8; ++c) v += hmp[(b * 8 + c) * HWP + t];
    }
    sv[t] = v; __syncthreads();
    if (t < 14) {                       // norm over H per column w (axis=1 of (H,W))
        float s = 0.f;
        for (int h = 0; h < 14; ++h) { float x = sv[h * 14 + t]; s += x * x; }
        nrm[t] = fmaxf(sqrtf(s), 1e-12f);
    }
    __syncthreads();
    if (t < HWP) sv[t] = sv[t] / nrm[t % 14];
    else         sv[t] = -INFINITY;
    si[t] = t;
    __syncthreads();
    // bitonic sort, descending with ties broken by lower index (jax top_k)
    for (int k = 2; k <= 256; k <<= 1) {
        for (int j = k >> 1; j > 0; j >>= 1) {
            const int ixj = t ^ j;
            if (ixj > t) {
                const bool desc = ((t & k) == 0);
                float va = sv[t], vb = sv[ixj];
                int   ia = si[t], ib = si[ixj];
                const bool aFirst = (va > vb) || (va == vb && ia < ib);
                if (desc ? !aFirst : aFirst) { sv[t] = vb; si[t] = ib; sv[ixj] = va; si[ixj] = ia; }
            }
            __syncthreads();
        }
    }
    if (t < SSEL) { pick_val[b * SSEL + t] = sv[t]; pick_pos[b * SSEL + t] = si[t]; }
}

// ---------- 4) gather pick_feat[b][d][s] = feat[b][d][pos] ----------
__global__ void k_gather(const float* __restrict__ feat, const int* __restrict__ pick_pos,
                         float* __restrict__ pf) {
    const int b = blockIdx.y, t = threadIdx.x;
    const int s = t & 31, d = blockIdx.x * 8 + (t >> 5);
    const int pos = pick_pos[b * SSEL + s];
    pf[((size_t)b * DIMC + d) * SSEL + s] = feat[((size_t)b * DIMC + d) * HWP + pos];
}

// ---------- 5) per-column l2 norms of bank[labels] and pick_feat ----------
__global__ void k_norms(const float* __restrict__ fb, const float* __restrict__ pf,
                        const int* __restrict__ labels, float* nbn, float* ncn) {
    const int b = blockIdx.x, t = threadIdx.x;
    const int s = t & 31, dp = t >> 5;
    const int l = labels[b];
    __shared__ float pB[8][32], pP[8][32];
    float aB = 0.f, aP = 0.f;
    for (int d = dp; d < DIMC; d += 8) {
        float xb = fb[((size_t)l * DIMC + d) * SSEL + s]; aB += xb * xb;
        float xp = pf[((size_t)b * DIMC + d) * SSEL + s]; aP += xp * xp;
    }
    pB[dp][s] = aB; pP[dp][s] = aP; __syncthreads();
    if (t < 32) {
        float sB = 0.f, sP = 0.f;
        for (int i = 0; i < 8; ++i) { sB += pB[i][t]; sP += pP[i][t]; }
        nbn[b * SSEL + t] = fmaxf(sqrtf(sB), 1e-8f);
        ncn[b * SSEL + t] = fmaxf(sqrtf(sP), 1e-8f);
    }
}

// ---------- 6) sim = nb^T * nc via v_wmma_f32_16x16x32_bf16 (bf16x3 split) ----------
// One wave per batch: 2x2 tiles of 16x16, K=2048 in 32-steps, 12 WMMAs/step.
__global__ void __launch_bounds__(32) k_sim_wmma(
        const float* __restrict__ fb, const float* __restrict__ pf,
        const int* __restrict__ labels, const float* __restrict__ nbn,
        const float* __restrict__ ncn, float* __restrict__ sim) {
    const int b = blockIdx.x;
    const int lane = threadIdx.x;
    const int l = labels[b];
    const bool hih = lane >= 16;
    const int mn = lane & 15;
    const float* A = fb + (size_t)l * (DIMC * SSEL);   // A[m][k] = bank[d0+k][a0+m]
    const float* B = pf + (size_t)b * (DIMC * SSEL);   // B[k][n] = pick [d0+k][c0+n]

    v8f acc00 = {}, acc01 = {}, acc10 = {}, acc11 = {};

    for (int d0 = 0; d0 < DIMC; d0 += 32) {
        v16bf a0h, a0l, a1h, a1l, b0h, b0l, b1h, b1l;
#pragma unroll
        for (int e = 0; e < 16; ++e) {
            // CDNA5 16-bit A fragment: K = e + (e>=8?8:0) + (lane>=16?8:0), M = lane&15
            const int kA = e + ((e >= 8) ? 8 : 0) + (hih ? 8 : 0);
            // CDNA5 16-bit B fragment: K = e + (e>=8?8:0) + (lane>=16?8:0), N = lane&15
            const int kB = e + ((e >= 8) ? 8 : 0) + (hih ? 8 : 0);
            float x0 = A[(size_t)(d0 + kA) * SSEL + mn];
            float x1 = A[(size_t)(d0 + kA) * SSEL + 16 + mn];
            float y0 = B[(size_t)(d0 + kB) * SSEL + mn];
            float y1 = B[(size_t)(d0 + kB) * SSEL + 16 + mn];
            __bf16 h;
            h = f2bf(x0); a0h[e] = h; a0l[e] = f2bf(x0 - bf2f(h));
            h = f2bf(x1); a1h[e] = h; a1l[e] = f2bf(x1 - bf2f(h));
            h = f2bf(y0); b0h[e] = h; b0l[e] = f2bf(y0 - bf2f(h));
            h = f2bf(y1); b1h[e] = h; b1l[e] = f2bf(y1 - bf2f(h));
        }
#define WM(C_, A_, B_) C_ = __builtin_amdgcn_wmma_f32_16x16x32_bf16(false, A_, false, B_, (short)0, C_, false, false)
        WM(acc00, a0h, b0h); WM(acc00, a0h, b0l); WM(acc00, a0l, b0h);
        WM(acc01, a0h, b1h); WM(acc01, a0h, b1l); WM(acc01, a0l, b1h);
        WM(acc10, a1h, b0h); WM(acc10, a1h, b0l); WM(acc10, a1l, b0h);
        WM(acc11, a1h, b1h); WM(acc11, a1h, b1l); WM(acc11, a1l, b1h);
#undef WM
    }
    // C/D layout: VGPR r -> M = r + (lane>=16?8:0), N = lane&15; fold in l2 norms
    const float* nb = nbn + b * SSEL;
    const float* nc = ncn + b * SSEL;
    float* o = sim + (size_t)b * (SSEL * SSEL);
#pragma unroll
    for (int r = 0; r < 8; ++r) {
        const int m = r + (hih ? 8 : 0);
        o[m * 32 + mn]               = acc00[r] / (nb[m]      * nc[mn]);
        o[m * 32 + 16 + mn]          = acc01[r] / (nb[m]      * nc[16 + mn]);
        o[(16 + m) * 32 + mn]        = acc10[r] / (nb[16 + m] * nc[mn]);
        o[(16 + m) * 32 + 16 + mn]   = acc11[r] / (nb[16 + m] * nc[16 + mn]);
    }
}

// ---------- 7) Sinkhorn (one wave per batch) + write otmaps ----------
__global__ void __launch_bounds__(32) k_sinkhorn(
        const float* __restrict__ sim, const float* __restrict__ bct,
        const float* __restrict__ pick_val, const int* __restrict__ labels,
        const int* __restrict__ fwdm, float* __restrict__ out_main) {
    const int b = blockIdx.x, t = threadIdx.x;
    const int l = labels[b];
    __shared__ float Ks[32][33];
    __shared__ float sh[32], us[32], vs[32];
    // _marg(bct[l])
    float wa = fmaxf(bct[l * SSEL + t], 0.f);
    sh[t] = wa; __syncthreads();
    float sa = 0.f; for (int j = 0; j < 32; ++j) sa += sh[j];
    const float a_t = (sa > 0.f) ? (wa / fmaxf(sa, 1e-8f)) : (1.f / 32.f);
    __syncthreads();
    // _marg(pick_val[b])
    float wb = fmaxf(pick_val[b * SSEL + t], 0.f);
    sh[t] = wb; __syncthreads();
    float sb = 0.f; for (int j = 0; j < 32; ++j) sb += sh[j];
    const float b_t = (sb > 0.f) ? (wb / fmaxf(sb, 1e-8f)) : (1.f / 32.f);

    float kr[32];
#pragma unroll
    for (int j = 0; j < 32; ++j) {
        float K = expf(sim[(size_t)b * 1024 + t * 32 + j] * 20.0f);  // /EPS_T
        kr[j] = K; Ks[t][j] = K;
    }
    vs[t] = 1.f;
    __syncthreads();
    float u = 1.f;
    for (int it = 0; it < 10; ++it) {
        float acc = 0.f;
#pragma unroll
        for (int j = 0; j < 32; ++j) acc += kr[j] * vs[j];
        u = a_t / fmaxf(acc, 1e-8f);
        us[t] = u; __syncthreads();
        float acc2 = 0.f;
#pragma unroll
        for (int i = 0; i < 32; ++i) acc2 += Ks[i][t] * us[i];
        vs[t] = b_t / fmaxf(acc2, 1e-8f);
        __syncthreads();
    }
    const float fm = fwdm[b] ? 1.f : 0.f;
    float* o = out_main + (size_t)b * 7168 + t * 32;
#pragma unroll
    for (int j = 0; j < 32; ++j) o[j] = u * kr[j] * vs[j] * fm;
}

// ---------- 8) per-class winner (duplicate-label scatter resolution) ----------
__global__ void k_winner(const int* __restrict__ labels, const int* __restrict__ updm,
                         int* __restrict__ winner) {
    const int t = threadIdx.x;
    if (t < NCLS) winner[t] = -1;
    __syncthreads();
    if (t < BSZ && updm[t]) atomicMax(&winner[labels[t]], t);
}

// ---------- 9) main extras: err_ct | bank_ct | err_bank_ct ----------
__global__ void k_main_extra(const float* __restrict__ pf, const float* __restrict__ ctx,
                             const int* __restrict__ labels, const int* __restrict__ pred_pos,
                             const int* __restrict__ errm, float* __restrict__ out_main) {
    const int b = blockIdx.y;
    const int t = blockIdx.x * 256 + threadIdx.x;   // 0..6143
    const int r = t >> 11, d = t & 2047;
    const float ef = errm[b] ? 1.f : 0.f;
    float v;
    if (r == 0)      v = pf[((size_t)b * DIMC + d) * SSEL] * ef;
    else if (r == 1) v = ctx[(size_t)labels[b] * DIMC + d] * ef;
    else             v = ctx[(size_t)pred_pos[b] * DIMC + d] * ef;
    out_main[(size_t)b * 7168 + 1024 + t] = v;
}

// ---------- 10) new_fb stream (52.7 MB) ----------
__global__ void k_new_fb(const float* __restrict__ fb, const float* __restrict__ pf,
                         const int* __restrict__ winner, float* __restrict__ out_fb) {
    const int gid = blockIdx.x * 256 + threadIdx.x;
    const int l = gid >> 16;          // / (2048*32)
    const int rem = gid & 65535;
    const int w = winner[l];
    out_fb[gid] = (w >= 0) ? pf[(size_t)w * 65536 + rem] : fb[gid];
}

// ---------- 11) new_ctx stream ----------
__global__ void k_new_ctx(const float* __restrict__ ctx, const float* __restrict__ pf,
                          const int* __restrict__ winner, float* __restrict__ out_ctx) {
    const int gid = blockIdx.x * 256 + threadIdx.x;   // < 201*2048
    const int l = gid >> 11, d = gid & 2047;
    const int w = winner[l];
    const float c = ctx[gid];
    out_ctx[gid] = (w >= 0) ? (0.8f * pf[((size_t)w * DIMC + d) * SSEL] + 0.2f * c) : c;
}

// ---------- 12) new_bct + new_bc ----------
__global__ void k_new_small(const float* __restrict__ bct, const float* __restrict__ bconf,
                            const float* __restrict__ pick_val, const float* __restrict__ pred_val,
                            const int* __restrict__ winner, float* __restrict__ out_bct,
                            float* __restrict__ out_bc) {
    const int gid = blockIdx.x * 256 + threadIdx.x;
    if (gid < NCLS * SSEL) {
        const int l = gid / SSEL, s = gid % SSEL;
        const int w = winner[l];
        out_bct[gid] = (w >= 0) ? pick_val[w * SSEL + s] : bct[gid];
    } else if (gid < NCLS * SSEL + NCLS) {
        const int l = gid - NCLS * SSEL;
        const int w = winner[l];
        out_bc[l] = (w >= 0) ? pred_val[w] : bconf[l];
    }
}

// ---------------------------- launch ----------------------------
extern "C" void kernel_launch(void* const* d_in, const int* in_sizes, int n_in,
                              void* d_out, int out_size, void* d_ws, size_t ws_size,
                              hipStream_t stream) {
    (void)in_sizes; (void)n_in; (void)out_size; (void)ws_size;
    const float* scores = (const float*)d_in[0];
    const float* feat   = (const float*)d_in[1];
    const float* fb     = (const float*)d_in[2];
    const float* bct    = (const float*)d_in[3];
    const float* bconf  = (const float*)d_in[4];
    const float* ctx    = (const float*)d_in[5];
    const int*   labels = (const int*)d_in[6];

    // workspace layout (4-byte units)
    float* wsf = (float*)d_ws;
    int*   wsi = (int*)d_ws;
    int*   pred_pos = wsi + 0;          // [128]
    float* pred_val = wsf + 128;        // [128]
    int*   updm     = wsi + 256;        // [128]
    int*   fwdm     = wsi + 384;        // [128]
    int*   errm     = wsi + 512;        // [128]
    int*   winner   = wsi + 640;        // [201] (pad to 896)
    int*   pick_pos = wsi + 896;        // [128*32]
    float* pick_val = wsf + 4992;       // [128*32]
    float* nbn      = wsf + 9088;       // [128*32]
    float* ncn      = wsf + 13184;      // [128*32]
    float* sim      = wsf + 17280;      // [128*1024]
    float* hmp      = wsf + 148352;     // [128*8*196]
    float* pf       = wsf + 349056;     // [128*2048*32]  (~33.5 MB)

    float* out      = (float*)d_out;
    float* out_main = out;              // 128*7168
    float* out_fb   = out + 917504;     // 201*2048*32
    float* out_bct  = out + 14090240;   // 201*32
    float* out_bc   = out + 14096672;   // 201
    float* out_ctx  = out + 14096873;   // 201*2048

    k_scores   <<<BSZ, 256, 0, stream>>>(scores, bconf, ctx, labels,
                                         pred_pos, pred_val, updm, fwdm, errm);
    k_hm_part  <<<dim3(8, BSZ), 256, 0, stream>>>(feat, hmp);
    k_topk     <<<BSZ, 256, 0, stream>>>(hmp, pick_val, pick_pos);
    k_gather   <<<dim3(256, BSZ), 256, 0, stream>>>(feat, pick_pos, pf);
    k_norms    <<<BSZ, 256, 0, stream>>>(fb, pf, labels, nbn, ncn);
    k_sim_wmma <<<BSZ, 32, 0, stream>>>(fb, pf, labels, nbn, ncn, sim);
    k_sinkhorn <<<BSZ, 32, 0, stream>>>(sim, bct, pick_val, labels, fwdm, out_main);
    k_winner   <<<1, 256, 0, stream>>>(labels, updm, winner);
    k_main_extra<<<dim3(24, BSZ), 256, 0, stream>>>(pf, ctx, labels, pred_pos, errm, out_main);
    k_new_fb   <<<51456, 256, 0, stream>>>(fb, pf, winner, out_fb);       // 201*65536/256
    k_new_ctx  <<<1608, 256, 0, stream>>>(ctx, pf, winner, out_ctx);      // 201*2048/256
    k_new_small<<<26, 256, 0, stream>>>(bct, bconf, pick_val, pred_val, winner,
                                        out_bct, out_bc);
}
